// PairwiseScorer_30897994727574
// MI455X (gfx1250) — matrix operand order repaired and checked by reading, not compiled, not run on hardware
//
#include <hip/hip_runtime.h>

// CDNA5 / gfx1250. wave32. WMMA f32_16x16x32_f16.
typedef __attribute__((ext_vector_type(16))) _Float16 v16h;
typedef __attribute__((ext_vector_type(8)))  float    v8f;

#define SB 2     // batch
#define SS 512   // sequence
#define SH 256   // hidden

// ---------------------------------------------------------------------------
// Stage 1: projI[row,k] = b1[k] + sum_h node[row,h]*W1[h,k]
//          projJ[row,k] =          sum_h node[row,h]*W1[H+h,k]
// 1024 blocks x 256 threads; each thread owns one output column k.
// ---------------------------------------------------------------------------
__global__ __launch_bounds__(256) void proj_kernel(
    const float* __restrict__ node, const float* __restrict__ W1,
    const float* __restrict__ b1,
    float* __restrict__ projI, float* __restrict__ projJ)
{
    __shared__ float srow[SH];
    const int row = blockIdx.x;        // b*S + s
    const int k   = threadIdx.x;       // 0..255
    srow[k] = node[(size_t)row * SH + k];
    __syncthreads();

    float ai = b1[k];   // fold b1 into projI
    float aj = 0.f;
#pragma unroll 4
    for (int h = 0; h < SH; ++h) {
        const float nf = srow[h];
        ai = fmaf(nf, W1[(size_t)h * SH + k], ai);
        aj = fmaf(nf, W1[(size_t)(SH + h) * SH + k], aj);
    }
    projI[(size_t)row * SH + k] = ai;
    projJ[(size_t)row * SH + k] = aj;
}

// ---------------------------------------------------------------------------
// Stage 1b: pack W2 [256,128] f32 -> f16 in this kernel's B-operand lane layout (contiguous K, matched by A).
// B is 32x16 (KxN) per (kk,nn) tile: lanes 0-15 hold col N=lane, K=0..15;
// lanes 16-31 hold col N=lane-16, K=16..31 (2 f16 per VGPR, contiguous K).
// Linear layout: w2p[((kk*8+nn)*32 + lane)*16 + e], e -> K = kk*32+half*16+e.
// ---------------------------------------------------------------------------
__global__ __launch_bounds__(256) void packW2_kernel(
    const float* __restrict__ W2, _Float16* __restrict__ w2p)
{
    const int t    = blockIdx.x * 256 + threadIdx.x;  // 0..2047
    const int lane = t & 31;
    const int tile = t >> 5;                          // 0..63 = kk*8+nn
    const int nn   = tile & 7;
    const int kk   = tile >> 3;
    const int half = lane >> 4;
    const int n    = nn * 16 + (lane & 15);
    _Float16* dst  = w2p + (size_t)t * 16;
#pragma unroll
    for (int e = 0; e < 16; ++e) {
        const int k = kk * 32 + half * 16 + e;
        dst[e] = (_Float16)W2[(size_t)k * 128 + n];
    }
}

// ---------------------------------------------------------------------------
// Stage 2: fused pairwise MLP. One wave = 16 pairs (fixed i, 16 consecutive j).
// Block = 8 waves = 128 j's for one i. Grid = B*S*(S/128) = 4096.
// ---------------------------------------------------------------------------
__global__ __launch_bounds__(256) void pair_kernel(
    const float* __restrict__ bp,   const float* __restrict__ W1,
    const float* __restrict__ b2,   const float* __restrict__ W3,
    const float* __restrict__ b3,
    const float* __restrict__ projI, const float* __restrict__ projJ,
    const _Float16* __restrict__ w2p, float* __restrict__ out)
{
    __shared__ _Float16 sB[64 * 32 * 16];   // 64 KB packed W2 (all kk,nn tiles)
    __shared__ float    sW1r[SH];           // w1row = W1[2H, :]

    const int tid = threadIdx.x;

    // cooperative LDS fill (4096 uint4 / 256 threads)
    {
        const uint4* src = (const uint4*)w2p;
        uint4*       dst = (uint4*)sB;
#pragma unroll
        for (int it = 0; it < 16; ++it)
            dst[it * 256 + tid] = src[it * 256 + tid];
        sW1r[tid] = W1[(size_t)2 * SH * SH + tid];
    }
    __syncthreads();

    const int bid = blockIdx.x;
    const int jb  = bid & 3;                // 4 j-blocks of 128
    const int i   = (bid >> 2) & (SS - 1);
    const int b   = bid >> 11;

    const int lane = tid & 31;
    const int wv   = tid >> 5;
    const int M    = lane & 15;             // pair-in-tile / col-in-tile
    const int half = lane >> 4;
    const int j0   = jb * 128 + wv * 16;
    const int j    = j0 + M;

    const float  p  = bp[((size_t)b * SS + i) * SS + j];
    const float* pi = projI + ((size_t)b * SS + i) * SH;
    const float* pj = projJ + ((size_t)b * SS + j) * SH;

    // per-lane channel constants for the epilogue (channel = nn*16 + M)
    float b2l[8], w3l[8];
#pragma unroll
    for (int nn = 0; nn < 8; ++nn) {
        const int ch = nn * 16 + M;
        b2l[nn] = b2[ch];
        w3l[nn] = W3[ch];
    }

    v8f acc[8] = {};   // 8 N-tiles x 8 VGPRs of f32 accumulators

    for (int kk = 0; kk < 8; ++kk) {
        // Build A (16x32 f16): this lane = row M; contiguous K (matches packed B):
        // lanes 0-15: K = kk*32 + {0..15}; lanes 16-31: +16.
        float hv[16];
#pragma unroll
        for (int g = 0; g < 2; ++g) {
            const int kb = kk * 32 + half * 16 + g * 8;
            const float4 a0 = *(const float4*)(pi + kb);
            const float4 a1 = *(const float4*)(pi + kb + 4);
            const float4 c0 = *(const float4*)(pj + kb);
            const float4 c1 = *(const float4*)(pj + kb + 4);
            const float4 r0 = *(const float4*)(sW1r + kb);
            const float4 r1 = *(const float4*)(sW1r + kb + 4);
            const float av[8] = {a0.x,a0.y,a0.z,a0.w,a1.x,a1.y,a1.z,a1.w};
            const float cv[8] = {c0.x,c0.y,c0.z,c0.w,c1.x,c1.y,c1.z,c1.w};
            const float rv[8] = {r0.x,r0.y,r0.z,r0.w,r1.x,r1.y,r1.z,r1.w};
#pragma unroll
            for (int e = 0; e < 8; ++e)
                hv[g * 8 + e] = fmaxf(fmaf(p, rv[e], av[e] + cv[e]), 0.f);
        }
        v16h A;
#pragma unroll
        for (int e = 0; e < 16; ++e) A[e] = (_Float16)hv[e];

        // 8 WMMAs, A reused across N-tiles; B from LDS (2x ds_load_b128 each)
#pragma unroll
        for (int nn = 0; nn < 8; ++nn) {
            const v16h Bv = *(const v16h*)(sB + ((size_t)(kk * 8 + nn) * 32 + lane) * 16);
            acc[nn] = __builtin_amdgcn_wmma_f32_16x16x32_f16(
                false, A, false, Bv, (short)0, acc[nn], false, false);
        }
    }

    // Epilogue: out[Mrow] = b3 + sum_ch relu(acc+b2[ch]) * W3[ch]
    // acc VGPR v: lanes 0-15 -> Mrow=v, lanes 16-31 -> Mrow=v+8; N = M.
    float part[8];
#pragma unroll
    for (int v = 0; v < 8; ++v) part[v] = 0.f;
#pragma unroll
    for (int nn = 0; nn < 8; ++nn) {
#pragma unroll
        for (int v = 0; v < 8; ++v)
            part[v] += fmaxf(acc[nn][v] + b2l[nn], 0.f) * w3l[nn];
    }

    const float b3v = b3[0];
#pragma unroll
    for (int v = 0; v < 8; ++v) {
        float s = part[v];
        s += __shfl_xor(s, 1, 32);   // reduce 16 channels within each half-wave
        s += __shfl_xor(s, 2, 32);
        s += __shfl_xor(s, 4, 32);
        s += __shfl_xor(s, 8, 32);
        if (M == 0) {
            const int jj = j0 + half * 8 + v;
            out[((size_t)b * SS + i) * SS + jj] = s + b3v;
        }
    }
}

// ---------------------------------------------------------------------------
extern "C" void kernel_launch(void* const* d_in, const int* in_sizes, int n_in,
                              void* d_out, int out_size, void* d_ws, size_t ws_size,
                              hipStream_t stream) {
    (void)in_sizes; (void)n_in; (void)out_size; (void)ws_size;
    const float* node = (const float*)d_in[0];   // [B,S,H]
    const float* bp   = (const float*)d_in[1];   // [B,S,S]
    const float* W1   = (const float*)d_in[2];   // [2H+1,H]
    const float* b1   = (const float*)d_in[3];   // [H]
    const float* W2   = (const float*)d_in[4];   // [H,H/2]
    const float* b2   = (const float*)d_in[5];   // [H/2]
    const float* W3   = (const float*)d_in[6];   // [H/2]
    const float* b3   = (const float*)d_in[7];   // [1]
    float* out = (float*)d_out;

    char* ws = (char*)d_ws;
    float*     projI = (float*)ws;                        // 1 MB
    float*     projJ = (float*)(ws + (1u << 20));         // 1 MB
    _Float16*  w2p   = (_Float16*)(ws + (2u << 20));      // 64 KB

    proj_kernel<<<SB * SS, 256, 0, stream>>>(node, W1, b1, projI, projJ);
    packW2_kernel<<<8, 256, 0, stream>>>(W2, w2p);
    pair_kernel<<<SB * SS * (SS / 128), 256, 0, stream>>>(
        bp, W1, b2, W3, b3, projI, projJ, w2p, out);
}
